// RelGatCatLayer_57836029608138
// MI455X (gfx1250) — matrix operand order repaired and checked by reading, not compiled, not run on hardware
//
#include <hip/hip_runtime.h>
#include <hip/hip_bf16.h>
#include <stdint.h>

// ---------------------------------------------------------------------------
// RelGAT-style layer for MI455X (gfx1250, wave32, WMMA bf16 w/ fp32 accum).
// Memory-bound (~1.3GB of E x 256 fp32 traffic -> ~56us @ 23.3TB/s):
// fp32 global traffic, bf16 only in registers/LDS for v_wmma, CSR-based
// segment softmax/aggregation (no global float atomics). This revision
// removes all predication from the WMMA K-loops (clamped row pointers,
// garbage rows/cols are never stored) and uses b128 global/LDS loads.
// ---------------------------------------------------------------------------

typedef __attribute__((ext_vector_type(16))) __bf16 v16bf;
typedef __attribute__((ext_vector_type(8)))  __bf16 v8bf;
typedef __attribute__((ext_vector_type(4)))  __bf16 v4bf;
typedef __attribute__((ext_vector_type(8)))  float  v8f;
typedef __attribute__((ext_vector_type(4)))  float  f32x4;

#define SLOPE (11.0f / 48.0f)

// ----------------------------- small utilities ------------------------------

__global__ void f32_to_bf16_kernel(const float* __restrict__ in,
                                   __bf16* __restrict__ out, int n) {
    int i = blockIdx.x * blockDim.x + threadIdx.x;
    if (i < n) out[i] = (__bf16)in[i];
}

__global__ void zero_i32_kernel(int* __restrict__ p, int n) {
    int i = blockIdx.x * blockDim.x + threadIdx.x;
    if (i < n) p[i] = 0;
}

__global__ void count_deg_kernel(const int* __restrict__ dst,
                                 int* __restrict__ cnt, int E) {
    int i = blockIdx.x * blockDim.x + threadIdx.x;
    if (i < E) atomicAdd(&cnt[dst[i]], 1);
}

// Single-block exclusive scan over cnt[0..n) -> off[0..n], off[0] = 0.
__global__ __launch_bounds__(1024) void scan_offsets_kernel(
    const int* __restrict__ cnt, int* __restrict__ off, int n) {
    __shared__ int buf[1024];
    __shared__ int carry;
    if (threadIdx.x == 0) carry = 0;
    __syncthreads();
    for (int base = 0; base < n; base += 1024) {
        int i = base + threadIdx.x;
        int v = (i < n) ? cnt[i] : 0;
        buf[threadIdx.x] = v;
        __syncthreads();
        for (int d = 1; d < 1024; d <<= 1) {
            int t = (threadIdx.x >= d) ? buf[threadIdx.x - d] : 0;
            __syncthreads();
            buf[threadIdx.x] += t;
            __syncthreads();
        }
        if (i < n) off[i + 1] = carry + buf[threadIdx.x];
        __syncthreads();
        if (threadIdx.x == 1023) carry += buf[1023];
        __syncthreads();
    }
    if (threadIdx.x == 0) off[0] = 0;
}

__global__ void scatter_edges_kernel(const int* __restrict__ dst,
                                     const int* __restrict__ off,
                                     int* __restrict__ cursor,
                                     int* __restrict__ elist, int E) {
    int i = blockIdx.x * blockDim.x + threadIdx.x;
    if (i < E) {
        int d = dst[i];
        int p = atomicAdd(&cursor[d], 1);
        elist[off[d] + p] = i;
    }
}

// ------------------------- fragment pack helpers ----------------------------

__device__ inline v16bf pack_a_from_f32(f32x4 a0, f32x4 a1, f32x4 a2, f32x4 a3) {
    v16bf a;
#pragma unroll
    for (int j = 0; j < 4; ++j) {
        a[j]      = (__bf16)a0[j];
        a[4 + j]  = (__bf16)a1[j];
        a[8 + j]  = (__bf16)a2[j];
        a[12 + j] = (__bf16)a3[j];
    }
    return a;
}

__device__ inline v16bf pack16(v8bf lo, v8bf hi) {
    v16bf r;
#pragma unroll
    for (int j = 0; j < 8; ++j) { r[j] = lo[j]; r[8 + j] = hi[j]; }
    return r;
}

// --------------------- generic WMMA GEMM: C = A @ W^T + b -------------------
// A: fp32 [M,K] (cvt to bf16 in regs), W: bf16 [F,K], C: fp32 [M,F].
// Optional epilogue gather-add: C[m,n] += addTab[idx[m]*addF + n].
// One wave per 16x16 output tile; blockIdx.x = rt*colTiles + ct.
// No predication in the K-loop: row/col pointers are clamped; out-of-range
// rows(A)/cols(B) only pollute C entries that the guarded store skips.

__global__ __launch_bounds__(32) void gemm_nt_wmma(
    const float* __restrict__ A, int M, int K,
    const __bf16* __restrict__ Wb, const float* __restrict__ bias, int F,
    const float* __restrict__ addTab, const int* __restrict__ idx, int addF,
    float* __restrict__ C, int colTiles) {
    int bx   = blockIdx.x;
    int ct   = bx % colTiles;
    int rt   = bx / colTiles;
    int lane = threadIdx.x & 31;
    int row0 = rt * 16;
    int col0 = ct * 16;

    int m  = row0 + (lane & 15);
    int n  = col0 + (lane & 15);
    int mc = min(m, M - 1);
    int nc = min(n, F - 1);

    // Per-lane base pointers (K-half select baked in).
    const float*  Arow = A  + (size_t)mc * K + (lane >> 4) * 8;
    const __bf16* Brow = Wb + (size_t)nc * K + (lane >> 4) * 8;

    v8f acc = {};
    for (int k0 = 0; k0 < K; k0 += 32) {
        f32x4 a0 = *(const f32x4*)(Arow + k0);
        f32x4 a1 = *(const f32x4*)(Arow + k0 + 4);
        f32x4 a2 = *(const f32x4*)(Arow + k0 + 16);
        f32x4 a3 = *(const f32x4*)(Arow + k0 + 20);
        v8bf  b0 = *(const v8bf*)(Brow + k0);
        v8bf  b1 = *(const v8bf*)(Brow + k0 + 16);
        v16bf a = pack_a_from_f32(a0, a1, a2, a3);
        v16bf b = pack16(b0, b1);
        acc = __builtin_amdgcn_wmma_f32_16x16x32_bf16(
            false, a, false, b, (short)0, acc, false, false);
    }
#pragma unroll
    for (int i = 0; i < 8; ++i) {
        int mm = row0 + i + (lane >> 4) * 8;
        if (mm < M && n < F) {
            float v = acc[i] + bias[n];
            if (idx) v += addTab[(size_t)idx[mm] * addF + n];
            C[(size_t)mm * F + n] = v;
        }
    }
}

// --------- fused edge message: msg = (nf[src] + ef) @ W1^T + b1  ------------
// 128 threads = 4 waves per block; 64 edges per block; LDS-staged bf16 X tile.
// Each wave owns a 16-row tile, preloads 8 A-fragments (full K=256) once via
// ds_load_b128 and reuses them across 16 column tiles (128 v_wmma per wave).

#define EB 64
#define XPITCH 264  // 256 + 8 bf16 pad; row stride 528B = 33*16 (16B aligned)

__global__ __launch_bounds__(128) void edge_msg_wmma(
    const float* __restrict__ nf, const float* __restrict__ ef,
    const int* __restrict__ src, const __bf16* __restrict__ W1b,
    const float* __restrict__ b1, float* __restrict__ msg, int E) {
    __shared__ __bf16 xt[EB * XPITCH];  // 33,792 B

    int e0 = blockIdx.x * EB;
    // Stage X = nf[src[e]] + ef[e] as bf16; float4 loads, b64 LDS stores.
    for (int t = threadIdx.x; t < EB * 64; t += blockDim.x) {
        int r  = t >> 6;          // row within tile
        int cq = (t & 63) * 4;    // column (quad)
        int e  = min(e0 + r, E - 1);
        int sv = src[e];
        f32x4 nv = *(const f32x4*)(nf + (size_t)sv * 256 + cq);
        f32x4 ev = *(const f32x4*)(ef + (size_t)e * 256 + cq);
        v4bf o;
#pragma unroll
        for (int j = 0; j < 4; ++j) o[j] = (__bf16)(nv[j] + ev[j]);
        *(v4bf*)(xt + r * XPITCH + cq) = o;
    }
    __syncthreads();

    int wave = threadIdx.x >> 5;
    int lane = threadIdx.x & 31;
    int r0   = wave * 16;  // row tile within block

    // Preload all 8 K-step A fragments for this wave's 16 rows (ds_load_b128).
    const __bf16* xrow = xt + (r0 + (lane & 15)) * XPITCH + (lane >> 4) * 8;
    v16bf afrag[8];
#pragma unroll
    for (int kk = 0; kk < 8; ++kk) {
        v8bf lo = *(const v8bf*)(xrow + kk * 32);
        v8bf hi = *(const v8bf*)(xrow + kk * 32 + 16);
        afrag[kk] = pack16(lo, hi);
    }

    const __bf16* Brow = W1b + (size_t)(lane & 15) * 256 + (lane >> 4) * 8;
    for (int ctile = 0; ctile < 16; ++ctile) {
        const __bf16* Bt = Brow + (size_t)ctile * 16 * 256;
        v8f acc = {};
#pragma unroll
        for (int kk = 0; kk < 8; ++kk) {
            v8bf b0 = *(const v8bf*)(Bt + kk * 32);
            v8bf b1 = *(const v8bf*)(Bt + kk * 32 + 16);
            acc = __builtin_amdgcn_wmma_f32_16x16x32_bf16(
                false, afrag[kk], false, pack16(b0, b1), (short)0, acc,
                false, false);
        }
        int n   = ctile * 16 + (lane & 15);
        float bn = b1[n];
#pragma unroll
        for (int i = 0; i < 8; ++i) {
            int m = e0 + r0 + i + (lane >> 4) * 8;
            if (m < E) msg[(size_t)m * 256 + n] = acc[i] + bn;
        }
    }
}

// ------------------- per-node softmax + weighted aggregation ----------------
// One 256-thread block per destination node; CSR edge list; everything in LDS.

__device__ inline unsigned f32_key(float v) {
    unsigned u = __float_as_uint(v);
    return (u & 0x80000000u) ? ~u : (u | 0x80000000u);
}
__device__ inline float key_f32(unsigned k) {
    return (k & 0x80000000u) ? __uint_as_float(k & 0x7fffffffu)
                             : __uint_as_float(~k);
}

__global__ __launch_bounds__(256) void node_aggregate(
    const float* __restrict__ msg, const float* __restrict__ s,
    const int* __restrict__ off, const int* __restrict__ elist,
    const float* __restrict__ self2, const float* __restrict__ self3,
    float* __restrict__ out, int Nn) {
    int node = blockIdx.x;
    int tid  = threadIdx.x;
    if (node >= Nn) return;

    __shared__ unsigned smax_k[8];
    __shared__ float    smax_f[8];
    __shared__ float    denom[8];
    __shared__ float    wbar_s[256];

    int beg = off[node];
    int deg = off[node + 1] - beg;
    float acc = 0.0f;

    if (deg > 0) {
        if (tid < 8) { smax_k[tid] = 0u; denom[tid] = 0.0f; }
        __syncthreads();
        for (int i = tid; i < deg; i += 256) {
            int e = elist[beg + i];
#pragma unroll
            for (int h = 0; h < 8; ++h)
                atomicMax(&smax_k[h], f32_key(s[(size_t)e * 8 + h]));
        }
        __syncthreads();
        if (tid < 8) smax_f[tid] = key_f32(smax_k[tid]);
        __syncthreads();
        for (int i = tid; i < deg; i += 256) {
            int e = elist[beg + i];
#pragma unroll
            for (int h = 0; h < 8; ++h)
                atomicAdd(&denom[h], __expf(s[(size_t)e * 8 + h] - smax_f[h]));
        }
        __syncthreads();
        // Chunked: stage per-edge head-mean weight, then coalesced column sum.
        for (int base = 0; base < deg; base += 256) {
            int chunk = min(256, deg - base);
            if (tid < chunk) {
                int e = elist[beg + base + tid];
                float wb = 0.0f;
#pragma unroll
                for (int h = 0; h < 8; ++h)
                    wb += __expf(s[(size_t)e * 8 + h] - smax_f[h]) / denom[h];
                wbar_s[tid] = wb * 0.125f;
            }
            __syncthreads();
            for (int j = 0; j < chunk; ++j) {
                int e = elist[beg + base + j];
                acc += wbar_s[j] * msg[(size_t)e * 256 + tid];  // coalesced
            }
            __syncthreads();
        }
    }

    const float* selfp = (deg == 0) ? self3 : self2;
    float z = selfp[(size_t)node * 256 + tid] + acc;
    out[(size_t)node * 256 + tid] = (z >= 0.0f) ? z : SLOPE * z;
}

// --------------------------------- launch -----------------------------------

extern "C" void kernel_launch(void* const* d_in, const int* in_sizes, int n_in,
                              void* d_out, int out_size, void* d_ws,
                              size_t ws_size, hipStream_t stream) {
    const float* nf = (const float*)d_in[0];
    const float* ef = (const float*)d_in[1];
    const float* W1 = (const float*)d_in[2];
    const float* b1 = (const float*)d_in[3];
    const float* W2 = (const float*)d_in[4];
    const float* b2 = (const float*)d_in[5];
    const float* W3 = (const float*)d_in[6];
    const float* b3 = (const float*)d_in[7];
    const float* W4 = (const float*)d_in[8];
    const float* b4 = (const float*)d_in[9];
    const float* W5 = (const float*)d_in[10];
    const float* b5 = (const float*)d_in[11];
    const int*   src = (const int*)d_in[12];
    const int*   dst = (const int*)d_in[13];
    float* out = (float*)d_out;

    const int D = 256, H = 256, NH = 8;
    const int N = in_sizes[0] / D;
    const int E = in_sizes[1] / D;

    // ---- workspace carve-out (256B aligned) ----
    size_t cur = 0;
    auto carve = [&](size_t bytes) -> char* {
        cur = (cur + 255) & ~(size_t)255;
        char* p = (char*)d_ws + cur;
        cur += bytes;
        return p;
    };
    __bf16* W1b = (__bf16*)carve((size_t)H * D * 2);
    __bf16* W2b = (__bf16*)carve((size_t)H * D * 2);
    __bf16* W3b = (__bf16*)carve((size_t)H * D * 2);
    __bf16* W4b = (__bf16*)carve((size_t)NH * H * 2);
    __bf16* W5b = (__bf16*)carve((size_t)NH * D * 2);
    float* msg    = (float*)carve((size_t)E * H * 4);
    float* sbuf   = (float*)carve((size_t)E * NH * 4);
    float* self2  = (float*)carve((size_t)N * H * 4);
    float* self3  = (float*)carve((size_t)N * H * 4);
    float* nproj5 = (float*)carve((size_t)N * NH * 4);
    int* cnt    = (int*)carve((size_t)N * 4);
    int* off    = (int*)carve((size_t)(N + 1) * 4);
    int* cursor = (int*)carve((size_t)N * 4);
    int* elist  = (int*)carve((size_t)E * 4);
    (void)ws_size;

    // 1) weights -> bf16 (stay hot in L2)
    f32_to_bf16_kernel<<<(H * D + 255) / 256, 256, 0, stream>>>(W1, W1b, H * D);
    f32_to_bf16_kernel<<<(H * D + 255) / 256, 256, 0, stream>>>(W2, W2b, H * D);
    f32_to_bf16_kernel<<<(H * D + 255) / 256, 256, 0, stream>>>(W3, W3b, H * D);
    f32_to_bf16_kernel<<<(NH * H + 255) / 256, 256, 0, stream>>>(W4, W4b, NH * H);
    f32_to_bf16_kernel<<<(NH * D + 255) / 256, 256, 0, stream>>>(W5, W5b, NH * D);

    // 2) node-side GEMMs (WMMA): self2, self3, nproj5
    int nrt = (N + 15) / 16;
    gemm_nt_wmma<<<nrt * (H / 16), 32, 0, stream>>>(
        nf, N, D, W2b, b2, H, nullptr, nullptr, 0, self2, H / 16);
    gemm_nt_wmma<<<nrt * (H / 16), 32, 0, stream>>>(
        nf, N, D, W3b, b3, H, nullptr, nullptr, 0, self3, H / 16);
    gemm_nt_wmma<<<nrt * 1, 32, 0, stream>>>(
        nf, N, D, W5b, b5, NH, nullptr, nullptr, 0, nproj5, 1);

    // 3) big fused edge GEMM: msg = (nf[src] + ef) @ W1^T + b1
    int nblk = (E + EB - 1) / EB;
    edge_msg_wmma<<<nblk, 128, 0, stream>>>(nf, ef, src, W1b, b1, msg, E);

    // 4) attention logits (WMMA): s = msg @ W4^T + b4 + nproj5[dst]
    int ert = (E + 15) / 16;
    gemm_nt_wmma<<<ert * 1, 32, 0, stream>>>(
        msg, E, H, W4b, b4, NH, nproj5, dst, NH, sbuf, 1);

    // 5) CSR build: counts -> exclusive scan -> scatter
    zero_i32_kernel<<<(N + 255) / 256, 256, 0, stream>>>(cnt, N);
    count_deg_kernel<<<(E + 255) / 256, 256, 0, stream>>>(dst, cnt, E);
    scan_offsets_kernel<<<1, 1024, 0, stream>>>(cnt, off, N);
    zero_i32_kernel<<<(N + 255) / 256, 256, 0, stream>>>(cursor, N);
    scatter_edges_kernel<<<(E + 255) / 256, 256, 0, stream>>>(dst, off, cursor,
                                                              elist, E);

    // 6) per-node softmax + weighted aggregation + self path + leaky-ReLU
    node_aggregate<<<N, 256, 0, stream>>>(msg, sbuf, off, elist, self2, self3,
                                          out, N);
}
